// GraphConvModel_20057497272865
// MI455X (gfx1250) — compile-verified, matrix operand order checked by reading, not run on hardware
//
#include <hip/hip_runtime.h>
#include <hip/hip_bf16.h>

typedef __attribute__((ext_vector_type(16))) __bf16 v16bf;
typedef __attribute__((ext_vector_type(8)))  float  v8f;

#define TILE_K 64
#define LN_EPS 1e-5f

static inline int cdiv(long a, long b) { return (int)((a + b - 1) / b); }

// ---------------------------------------------------------------- utilities
__global__ void k_zero(float* __restrict__ p, long n) {
  long i = (long)blockIdx.x * blockDim.x + threadIdx.x;
  if (i < n) p[i] = 0.0f;
}

__global__ void k_deg(const int* __restrict__ src, const int* __restrict__ dst,
                      float* __restrict__ dout, float* __restrict__ din, int E) {
  int e = blockIdx.x * blockDim.x + threadIdx.x;
  if (e >= E) return;
  atomicAdd(&dout[src[e]], 1.0f);
  atomicAdd(&din[dst[e]], 1.0f);
}

__global__ void k_rsqrt(float* __restrict__ d, int n) {
  int i = blockIdx.x * blockDim.x + threadIdx.x;
  if (i < n) d[i] = rsqrtf(fmaxf(d[i], 1.0f));
}

// ------------------------------------------------- WMMA GEMM: C = (A*rs) @ W
// A: Nrows x K (f32, row major), rs: per-row scale, W: K x 128 (f32), C: Nrows x 128.
// block = 256 threads (8 waves); block tile = 128 rows; wave tile = 16 rows x 128 cols.
//
// W is staged in LDS PRE-SWIZZLED into the hardware B-fragment layout so each lane's
// 16-element bf16 fragment is one contiguous 32-byte LDS read (2x ds_load_b128).
// All 8 B fragments of a 32-k half are preloaded into distinct registers before the
// WMMA chain so the DScnt waits form a descending ladder overlapping the WMMAs.
//
// B fragment layout (16-bit, 32x16): lane = kh*16 + lh, n = nt*16 + lh,
//   element e -> k_local = half*32 + 8*kh + (e<8 ? e : e+8)
//   => elements {0..7} and {8..15} are two contiguous 8-wide k-runs.
__global__ __launch_bounds__(256, 1)
void k_gemm_rs_wmma(const float* __restrict__ A, const float* __restrict__ rs,
                    const float* __restrict__ W, float* __restrict__ C,
                    int Nrows, int K) {
  // [half(2)][nt(8)][lane(32)][e(16)] bf16 = 16 KB
  __shared__ __bf16 sW[2 * 8 * 32 * 16];
  const int tid  = threadIdx.x;
  const int wave = tid >> 5;
  const int lane = tid & 31;
  const int lh   = lane & 15;   // m (A rows) / n (B cols) within fragment
  const int kh   = lane >> 4;   // k-half selector
  const int rowbase = blockIdx.x * 128 + wave * 16;

  const int  arow   = rowbase + lh;
  const int  arow_c = arow < Nrows ? arow : (Nrows - 1);
  const float scale = rs ? rs[arow_c] : 1.0f;
  const float* __restrict__ Arow = A + (long)arow_c * K;

  // per-thread W staging piece decode (4 pieces of 8 consecutive k values each)
  int pn[4], pk0[4];  // column n and local-k run start for each piece
#pragma unroll
  for (int q = 0; q < 4; ++q) {
    int p    = tid + q * 256;
    int run  = p & 1;
    int ln   = (p >> 1) & 31;
    int nt   = (p >> 6) & 7;
    int half = (p >> 9) & 1;
    pn[q]  = nt * 16 + (ln & 15);
    pk0[q] = half * 32 + 8 * (ln >> 4) + run * 16;
  }

  v8f acc[8];
#pragma unroll
  for (int nt = 0; nt < 8; ++nt)
#pragma unroll
    for (int j = 0; j < 8; ++j) acc[nt][j] = 0.0f;

  const v16bf* __restrict__ fr = (const v16bf*)sW;

  for (int kc = 0; kc < K; kc += TILE_K) {
    const bool full = (kc + TILE_K <= K);   // uniform per block

    // ---- stage W chunk, swizzled to fragment layout (uniform fast path)
    if (full) {
#pragma unroll
      for (int q = 0; q < 4; ++q) {
        const float* base = W + (long)(kc + pk0[q]) * 128 + pn[q];
        alignas(16) __bf16 t[8];
#pragma unroll
        for (int i = 0; i < 8; ++i) t[i] = (__bf16)base[(long)i * 128];
        *(float4*)&sW[(size_t)(tid + q * 256) * 8] = *(const float4*)t;
      }
    } else {
#pragma unroll
      for (int q = 0; q < 4; ++q) {
        alignas(16) __bf16 t[8];
#pragma unroll
        for (int i = 0; i < 8; ++i) {
          int gk = kc + pk0[q] + i;
          t[i] = (__bf16)((gk < K) ? W[(long)gk * 128 + pn[q]] : 0.0f);
        }
        *(float4*)&sW[(size_t)(tid + q * 256) * 8] = *(const float4*)t;
      }
    }
    __syncthreads();

    // ---- A for whole chunk: 4 runs of 8 consecutive floats
    float av[4][8];                        // [half*2+run][i]
    if (full) {                            // all runs in-bounds (K % 4 == 0 => aligned)
#pragma unroll
      for (int hr = 0; hr < 4; ++hr) {
        int ks = kc + (hr >> 1) * 32 + (hr & 1) * 16 + 8 * kh;
        float4 x0 = *(const float4*)(Arow + ks);
        float4 x1 = *(const float4*)(Arow + ks + 4);
        av[hr][0]=x0.x; av[hr][1]=x0.y; av[hr][2]=x0.z; av[hr][3]=x0.w;
        av[hr][4]=x1.x; av[hr][5]=x1.y; av[hr][6]=x1.z; av[hr][7]=x1.w;
      }
    } else {
#pragma unroll
      for (int hr = 0; hr < 4; ++hr) {
        int ks = kc + (hr >> 1) * 32 + (hr & 1) * 16 + 8 * kh;
#pragma unroll
        for (int i = 0; i < 8; ++i) {
          int k = ks + i;
          av[hr][i] = (k < K) ? Arow[k] : 0.0f;
        }
      }
    }

    // ---- WMMA over the two 32-k halves of this chunk
#pragma unroll
    for (int half = 0; half < 2; ++half) {
      int kbase = kc + half * 32;
      if (kbase < K) {
        // build A fragment first: its VALU work overlaps the B LDS loads below
        v16bf afrag;
#pragma unroll
        for (int e = 0; e < 16; ++e) {
          float v = (e < 8) ? av[half * 2 + 0][e] : av[half * 2 + 1][e - 8];
          afrag[e] = (__bf16)(v * scale);
        }
        // preload ALL 8 B fragments (distinct registers) -> 16 back-to-back
        // ds_load_b128, then a descending s_wait_dscnt ladder across the WMMAs
        const v16bf* hb = fr + (half * 8) * 32 + lane;
        v16bf b0 = hb[0 * 32], b1 = hb[1 * 32], b2 = hb[2 * 32], b3 = hb[3 * 32];
        v16bf b4 = hb[4 * 32], b5 = hb[5 * 32], b6 = hb[6 * 32], b7 = hb[7 * 32];
        acc[0] = __builtin_amdgcn_wmma_f32_16x16x32_bf16(false, afrag, false, b0, (short)0, acc[0], false, false);
        acc[1] = __builtin_amdgcn_wmma_f32_16x16x32_bf16(false, afrag, false, b1, (short)0, acc[1], false, false);
        acc[2] = __builtin_amdgcn_wmma_f32_16x16x32_bf16(false, afrag, false, b2, (short)0, acc[2], false, false);
        acc[3] = __builtin_amdgcn_wmma_f32_16x16x32_bf16(false, afrag, false, b3, (short)0, acc[3], false, false);
        acc[4] = __builtin_amdgcn_wmma_f32_16x16x32_bf16(false, afrag, false, b4, (short)0, acc[4], false, false);
        acc[5] = __builtin_amdgcn_wmma_f32_16x16x32_bf16(false, afrag, false, b5, (short)0, acc[5], false, false);
        acc[6] = __builtin_amdgcn_wmma_f32_16x16x32_bf16(false, afrag, false, b6, (short)0, acc[6], false, false);
        acc[7] = __builtin_amdgcn_wmma_f32_16x16x32_bf16(false, afrag, false, b7, (short)0, acc[7], false, false);
      }
    }
    __syncthreads();
  }

  // C/D layout: VGPR r -> row r (lanes 0-15) / row r+8 (lanes 16-31), col = lane%16
  if (rowbase < Nrows) {
#pragma unroll
    for (int nt = 0; nt < 8; ++nt)
#pragma unroll
      for (int r = 0; r < 8; ++r) {
        int row = rowbase + r + 8 * kh;
        int col = nt * 16 + lh;
        if (row < Nrows) C[(long)row * 128 + col] = acc[nt][r];
      }
  }
}

// ----------------------------------------------------------- edge scatter
__global__ void k_scatter(const float* __restrict__ xs, const int* __restrict__ src,
                          const int* __restrict__ dst, float* __restrict__ agg, int E) {
  long idx = (long)blockIdx.x * blockDim.x + threadIdx.x;
  if (idx >= (long)E * 32) return;
  int e  = (int)(idx >> 5);
  int f4 = (int)(idx & 31) << 2;
  const float4 v = *(const float4*)(xs + (long)src[e] * 128 + f4);
  float* b = agg + (long)dst[e] * 128 + f4;
  atomicAdd(b + 0, v.x); atomicAdd(b + 1, v.y);
  atomicAdd(b + 2, v.z); atomicAdd(b + 3, v.w);
}

// ---------------------------- fused  agg*rsqrt(deg_in)+b -> LayerNorm -> ELU
__global__ __launch_bounds__(256)
void k_ln_elu(const float* __restrict__ agg, const float* __restrict__ rsin,
              const float* __restrict__ bias, const float* __restrict__ g,
              const float* __restrict__ b, float* __restrict__ h, int N) {
  int wave = threadIdx.x >> 5, lane = threadIdx.x & 31;
  int row = blockIdx.x * 8 + wave;
  if (row >= N) return;
  float rs = rsin[row];
  int f0 = lane * 4;
  const float4 av = *(const float4*)(agg + (long)row * 128 + f0);
  const float4 bv = *(const float4*)(bias + f0);
  float x0 = av.x * rs + bv.x, x1 = av.y * rs + bv.y;
  float x2 = av.z * rs + bv.z, x3 = av.w * rs + bv.w;
  float s  = x0 + x1 + x2 + x3;
  float sq = x0*x0 + x1*x1 + x2*x2 + x3*x3;
#pragma unroll
  for (int m = 16; m >= 1; m >>= 1) {
    s  += __shfl_xor(s,  m, 32);
    sq += __shfl_xor(sq, m, 32);
  }
  float mu  = s * (1.0f / 128.0f);
  float var = sq * (1.0f / 128.0f) - mu * mu;
  float inv = rsqrtf(var + LN_EPS);
  const float4 gv  = *(const float4*)(g + f0);
  const float4 bbv = *(const float4*)(b + f0);
  float y0 = (x0 - mu) * inv * gv.x + bbv.x;
  float y1 = (x1 - mu) * inv * gv.y + bbv.y;
  float y2 = (x2 - mu) * inv * gv.z + bbv.z;
  float y3 = (x3 - mu) * inv * gv.w + bbv.w;
  float4 o;
  o.x = y0 > 0.0f ? y0 : expf(y0) - 1.0f;
  o.y = y1 > 0.0f ? y1 : expf(y1) - 1.0f;
  o.z = y2 > 0.0f ? y2 : expf(y2) - 1.0f;
  o.w = y3 > 0.0f ? y3 : expf(y3) - 1.0f;
  *(float4*)(h + (long)row * 128 + f0) = o;
}

// ----------------------------------------------------------- pooling + head
__global__ void k_seg_sum(const float* __restrict__ h, const int* __restrict__ gid,
                          float* __restrict__ sum, int N) {
  long idx = (long)blockIdx.x * blockDim.x + threadIdx.x;
  if (idx >= (long)N * 32) return;
  int node = (int)(idx >> 5);
  int f4 = (int)(idx & 31) << 2;
  int gg = gid[node];
  const float4 v = *(const float4*)(h + (long)node * 128 + f4);
  float* b = sum + (long)gg * 128 + f4;
  atomicAdd(b + 0, v.x); atomicAdd(b + 1, v.y);
  atomicAdd(b + 2, v.z); atomicAdd(b + 3, v.w);
}

__global__ void k_seg_cnt(const int* __restrict__ gid, float* __restrict__ cnt, int N) {
  int i = blockIdx.x * blockDim.x + threadIdx.x;
  if (i < N) atomicAdd(&cnt[gid[i]], 1.0f);
}

__global__ __launch_bounds__(128)
void k_head(const float* __restrict__ suml, const float* __restrict__ cntl,
            const float* __restrict__ sumn, const float* __restrict__ cntn,
            const float* __restrict__ Wfc, const float* __restrict__ bfc,
            const float* __restrict__ Wout, const float* __restrict__ bout,
            float* __restrict__ out) {
  __shared__ float hg[128];
  __shared__ float fc[128];
  int g = blockIdx.x, j = threadIdx.x;
  float cl = fmaxf(cntl[g], 1.0f), cn = fmaxf(cntn[g], 1.0f);
  hg[j] = suml[(long)g * 128 + j] / cl + sumn[(long)g * 128 + j] / cn;
  __syncthreads();
  float acc = bfc[j];
  for (int i = 0; i < 128; ++i) acc += hg[i] * Wfc[i * 128 + j];
  fc[j] = fmaxf(acc, 0.0f) * Wout[j];
  __syncthreads();
  for (int s = 64; s >= 1; s >>= 1) {
    if (j < s) fc[j] += fc[j + s];
    __syncthreads();
  }
  if (j == 0) out[g] = fc[0] + bout[0];
}

// ================================================================= driver
extern "C" void kernel_launch(void* const* d_in, const int* in_sizes, int n_in,
                              void* d_out, int out_size, void* d_ws, size_t ws_size,
                              hipStream_t stream) {
  const float* feat_l = (const float*)d_in[0];
  const float* feat_n = (const float*)d_in[1];
  const float* W0_l2n = (const float*)d_in[2];
  const float* b0_l2n = (const float*)d_in[3];
  const float* W0_n2l = (const float*)d_in[4];
  const float* b0_n2l = (const float*)d_in[5];
  const float* W_l2n  = (const float*)d_in[6];
  const float* b_l2n  = (const float*)d_in[7];
  const float* W_n2l  = (const float*)d_in[8];
  const float* b_n2l  = (const float*)d_in[9];
  const float* ln_g_n = (const float*)d_in[10];
  const float* ln_b_n = (const float*)d_in[11];
  const float* ln_g_l = (const float*)d_in[12];
  const float* ln_b_l = (const float*)d_in[13];
  const float* W_fc   = (const float*)d_in[14];
  const float* b_fc   = (const float*)d_in[15];
  const float* W_out  = (const float*)d_in[16];
  const float* b_out  = (const float*)d_in[17];
  const int* src_l2n  = (const int*)d_in[18];
  const int* dst_l2n  = (const int*)d_in[19];
  const int* src_n2l  = (const int*)d_in[20];
  const int* dst_n2l  = (const int*)d_in[21];
  const int* gid_l    = (const int*)d_in[22];
  const int* gid_n    = (const int*)d_in[23];

  const int N_L = in_sizes[0] / 300;
  const int N_N = in_sizes[1] / 200;
  const int E   = in_sizes[18];
  const int G   = out_size;            // (G,1) f32 output
  const int D   = 128;

  // ---- workspace carve-up
  float* p = (float*)d_ws;
  float* rs_out_l = p; p += N_L;
  float* rs_in_l  = p; p += N_L;
  float* rs_out_n = p; p += N_N;
  float* rs_in_n  = p; p += N_N;
  float* h_l   = p; p += (size_t)N_L * D;
  float* h_n   = p; p += (size_t)N_N * D;
  float* xs    = p; p += (size_t)(N_L > N_N ? N_L : N_N) * D;
  float* agg_l = p; p += (size_t)N_L * D;
  float* agg_n = p; p += (size_t)N_N * D;
  float* sum_l = p; p += (size_t)G * D;
  float* sum_n = p; p += (size_t)G * D;
  float* cnt_l = p; p += G;
  float* cnt_n = p; p += G;

  const int ZB = 256;

  // ---- degree -> rsqrt factors
  k_zero<<<cdiv(N_L, ZB), ZB, 0, stream>>>(rs_out_l, N_L);
  k_zero<<<cdiv(N_L, ZB), ZB, 0, stream>>>(rs_in_l,  N_L);
  k_zero<<<cdiv(N_N, ZB), ZB, 0, stream>>>(rs_out_n, N_N);
  k_zero<<<cdiv(N_N, ZB), ZB, 0, stream>>>(rs_in_n,  N_N);
  k_deg<<<cdiv(E, ZB), ZB, 0, stream>>>(src_l2n, dst_l2n, rs_out_l, rs_in_n, E);
  k_deg<<<cdiv(E, ZB), ZB, 0, stream>>>(src_n2l, dst_n2l, rs_out_n, rs_in_l, E);
  k_rsqrt<<<cdiv(N_L, ZB), ZB, 0, stream>>>(rs_out_l, N_L);
  k_rsqrt<<<cdiv(N_L, ZB), ZB, 0, stream>>>(rs_in_l,  N_L);
  k_rsqrt<<<cdiv(N_N, ZB), ZB, 0, stream>>>(rs_out_n, N_N);
  k_rsqrt<<<cdiv(N_N, ZB), ZB, 0, stream>>>(rs_in_n,  N_N);

  // ---- 3 bipartite GraphConv layers
  for (int i = 0; i < 3; ++i) {
    const float* Al = (i == 0) ? feat_l : h_l;
    const float* An = (i == 0) ? feat_n : h_n;
    const int Kl = (i == 0) ? 300 : 128;
    const int Kn = (i == 0) ? 200 : 128;
    const float* Wl = (i == 0) ? W0_l2n : (W_l2n + (size_t)(i - 1) * D * D);
    const float* Wn = (i == 0) ? W0_n2l : (W_n2l + (size_t)(i - 1) * D * D);
    const float* bl = (i == 0) ? b0_l2n : (b_l2n + (size_t)(i - 1) * D);
    const float* bn = (i == 0) ? b0_n2l : (b_n2l + (size_t)(i - 1) * D);

    // l -> n  (read h_l, accumulate into agg_n)
    k_gemm_rs_wmma<<<cdiv(N_L, 128), 256, 0, stream>>>(Al, rs_out_l, Wl, xs, N_L, Kl);
    k_zero<<<cdiv((long)N_N * D, ZB), ZB, 0, stream>>>(agg_n, (long)N_N * D);
    k_scatter<<<cdiv((long)E * 32, ZB), ZB, 0, stream>>>(xs, src_l2n, dst_l2n, agg_n, E);

    // n -> l  (read h_n before it is overwritten)
    k_gemm_rs_wmma<<<cdiv(N_N, 128), 256, 0, stream>>>(An, rs_out_n, Wn, xs, N_N, Kn);
    k_zero<<<cdiv((long)N_L * D, ZB), ZB, 0, stream>>>(agg_l, (long)N_L * D);
    k_scatter<<<cdiv((long)E * 32, ZB), ZB, 0, stream>>>(xs, src_n2l, dst_n2l, agg_l, E);

    // normalize + LN + ELU  -> new h_n, h_l
    k_ln_elu<<<cdiv(N_N, 8), 256, 0, stream>>>(agg_n, rs_in_n, bl,
        ln_g_n + (size_t)i * D, ln_b_n + (size_t)i * D, h_n, N_N);
    k_ln_elu<<<cdiv(N_L, 8), 256, 0, stream>>>(agg_l, rs_in_l, bn,
        ln_g_l + (size_t)i * D, ln_b_l + (size_t)i * D, h_l, N_L);
  }

  // ---- segment-mean pooling + MLP head
  k_zero<<<cdiv((long)G * D, ZB), ZB, 0, stream>>>(sum_l, (long)G * D);
  k_zero<<<cdiv((long)G * D, ZB), ZB, 0, stream>>>(sum_n, (long)G * D);
  k_zero<<<cdiv(G, ZB), ZB, 0, stream>>>(cnt_l, G);
  k_zero<<<cdiv(G, ZB), ZB, 0, stream>>>(cnt_n, G);
  k_seg_sum<<<cdiv((long)N_L * 32, ZB), ZB, 0, stream>>>(h_l, gid_l, sum_l, N_L);
  k_seg_sum<<<cdiv((long)N_N * 32, ZB), ZB, 0, stream>>>(h_n, gid_n, sum_n, N_N);
  k_seg_cnt<<<cdiv(N_L, ZB), ZB, 0, stream>>>(gid_l, cnt_l, N_L);
  k_seg_cnt<<<cdiv(N_N, ZB), ZB, 0, stream>>>(gid_n, cnt_n, N_N);
  k_head<<<G, 128, 0, stream>>>(sum_l, cnt_l, sum_n, cnt_n,
                                W_fc, b_fc, W_out, b_out, (float*)d_out);
}
